// TestMamba2Block_50929722196038
// MI455X (gfx1250) — matrix operand order repaired and checked by reading, not compiled, not run on hardware
//
#include <hip/hip_runtime.h>
#include <math.h>

// =====================================================================
// TestMamba2Block on MI455X (gfx1250, wave32, WMMA + async LDS DMA)
//
//   state = tanh(X @ B^T)        X:[16384,2048] B:[1024,2048]
//   out   = state @ C^T + D*X    C:[2048,1024]
//
// 137 GFLOP, ~284 MB min HBM traffic (~12us floor @ 23.3 TB/s).
// Compute: v_wmma_f32_16x16x32_bf16 (fp32 accumulate).
// Pass 0 packs X/B/C to bf16 once; GEMM hot loops stage tiles with
// GLOBAL_LOAD_ASYNC_TO_LDS_B128 (ASYNCcnt) into double-buffered LDS,
// 8 WMMAs per wave per K-step. d_ws: xh(64MB)|Bh(4MB)|Ch(4MB)|state(32MB).
// =====================================================================

typedef __bf16 bf16_t;
typedef __attribute__((ext_vector_type(16))) __bf16 v16bf;
typedef __attribute__((ext_vector_type(8)))  float  v8f;
typedef __attribute__((ext_vector_type(4)))  int    v4i;

#define AS1 __attribute__((address_space(1)))
#define AS3 __attribute__((address_space(3)))

#if __has_builtin(__builtin_amdgcn_global_load_async_to_lds_b128)
#define USE_ASYNC_LDS 1
#else
#define USE_ASYNC_LDS 0
#endif

#define DM   2048             // d_model
#define DSZ  1024             // d_state
#define MTOT 16384            // batch * seq

#define BM   128
#define BN   128
#define BK   32
#define LDSS 40               // BK + 8 pad: 80B row stride, conflict-free

__device__ __forceinline__ bf16_t f2bf(float f) {
    union { float f; unsigned u; } c; c.f = f;
    unsigned r = (c.u + 0x7FFFu + ((c.u >> 16) & 1u)) >> 16;  // RNE
    union { unsigned short s; bf16_t b; } o; o.s = (unsigned short)r;
    return o.b;
}

__device__ __forceinline__ void wait_async0() {
#if __has_builtin(__builtin_amdgcn_s_wait_asynccnt)
    __builtin_amdgcn_s_wait_asynccnt(0);
#else
    asm volatile("s_wait_asynccnt 0" ::: "memory");
#endif
}

// 16-byte async DMA global -> LDS (no VGPR round trip, ASYNCcnt tracked)
__device__ __forceinline__ void async_copy16(const bf16_t* g, bf16_t* l) {
#if USE_ASYNC_LDS
    __builtin_amdgcn_global_load_async_to_lds_b128(
        (AS1 v4i*)g, (AS3 v4i*)l, 0, 0);
#endif
}

// A fragment 16x32 bf16 (contiguous k order, not the ISA table order):
// lanes 0-15: M=0..15, VGPR0-7 K=0..15 (packed pairs)
// lanes16-31: same rows,        K=16..31
__device__ __forceinline__ v16bf load_a_frag(const bf16_t* sA, int row_base) {
    const int lane  = threadIdx.x & 31;
    const int m     = row_base + (lane & 15);
    const int khalf = (lane >> 4) * 16;
    const bf16_t* p = sA + m * LDSS;
    v16bf a;
#pragma unroll
    for (int v = 0; v < 8; ++v) {
        int k = 2 * v + khalf;
        a[2 * v]     = p[k];
        a[2 * v + 1] = p[k + 1];
    }
    return a;
}

// B fragment 32x16 bf16: lane%16 = column N; lanes0-15 K=0..15,
// lanes16-31 K=16..31; VGPR v packs K = khi+2v, khi+2v+1. sB holds sB[n][k].
__device__ __forceinline__ v16bf load_b_frag(const bf16_t* sB, int col_base) {
    const int lane = threadIdx.x & 31;
    const int n    = col_base + (lane & 15);
    const int khi  = (lane >> 4) * 16;
    const bf16_t* p = sB + n * LDSS + khi;
    v16bf b;
#pragma unroll
    for (int v = 0; v < 8; ++v) {
        b[2 * v]     = p[2 * v];
        b[2 * v + 1] = p[2 * v + 1];
    }
    return b;
}

// ----------------------- pack fp32 -> bf16 -------------------------------
__global__ __launch_bounds__(256)
void k_pack_bf16(const float* __restrict__ src, bf16_t* __restrict__ dst, size_t n) {
    const size_t i = ((size_t)blockIdx.x * 256 + threadIdx.x) * 4;
    if (i + 3 >= n) {
        if (i >= n) return;
        for (size_t j = i; j < n; ++j) dst[j] = f2bf(src[j]);
        return;
    }
    const float4 v = *(const float4*)(src + i);
    bf16_t* d = dst + i;
    d[0] = f2bf(v.x); d[1] = f2bf(v.y); d[2] = f2bf(v.z); d[3] = f2bf(v.w);
}

// --------------- shared double-buffered WMMA main loop -------------------
// BM=BN=128, 8 waves; each wave owns 2 M-tiles x 4 N-tiles -> 8 WMMA/K-step.
template <int KD>
__device__ __forceinline__ void gemm_core(const bf16_t* __restrict__ A,
                                          const bf16_t* __restrict__ Bmat,
                                          int m0, int n0,
                                          bf16_t* sA, bf16_t* sB,
                                          v8f acc[2][4]) {
    const int t    = threadIdx.x;
    const int wave = t >> 5;
    const int wm2  = (wave & 3) * 2;      // M tiles wm2, wm2+1
    const int wn4  = (wave >> 2) * 4;     // N tiles wn4..wn4+3

    const int row = t >> 1;               // 0..127
    const int sl  = t & 1;                // slots {sl, sl+2} of 4 x 16B per row

    const bf16_t* gA = A    + (size_t)(m0 + row) * KD + sl * 8;
    const bf16_t* gB = Bmat + (size_t)(n0 + row) * KD + sl * 8;
    bf16_t* lA = sA + row * LDSS + sl * 8;
    bf16_t* lB = sB + row * LDSS + sl * 8;
    const int bufStride = BM * LDSS;      // elements per buffer

#if USE_ASYNC_LDS
    auto stage = [&](int buf, int k0) {   // 4x global_load_async_to_lds_b128
        bf16_t* la = lA + buf * bufStride;
        bf16_t* lb = lB + buf * bufStride;
        async_copy16(gA + k0,      la);
        async_copy16(gA + k0 + 16, la + 16);
        async_copy16(gB + k0,      lb);
        async_copy16(gB + k0 + 16, lb + 16);
    };
#else
    uint4 ra0, ra1, rb0, rb1;
    auto g_load = [&](int k0) {
        ra0 = *(const uint4*)(gA + k0);
        ra1 = *(const uint4*)(gA + k0 + 16);
        rb0 = *(const uint4*)(gB + k0);
        rb1 = *(const uint4*)(gB + k0 + 16);
    };
    auto l_store = [&](int buf) {
        bf16_t* la = lA + buf * bufStride;
        bf16_t* lb = lB + buf * bufStride;
        *(uint4*)(la)      = ra0;
        *(uint4*)(la + 16) = ra1;
        *(uint4*)(lb)      = rb0;
        *(uint4*)(lb + 16) = rb1;
    };
#endif

    auto compute = [&](int buf) {         // 6 frag loads, 8 WMMAs
        const bf16_t* pA = sA + buf * bufStride;
        const bf16_t* pB = sB + buf * bufStride;
        v16bf a0 = load_a_frag(pA, (wm2 + 0) * 16);
        v16bf a1 = load_a_frag(pA, (wm2 + 1) * 16);
#pragma unroll
        for (int j = 0; j < 4; ++j) {
            v16bf b = load_b_frag(pB, (wn4 + j) * 16);
            acc[0][j] = __builtin_amdgcn_wmma_f32_16x16x32_bf16(false, a0, false, b, (short)0, acc[0][j], false, false);
            acc[1][j] = __builtin_amdgcn_wmma_f32_16x16x32_bf16(false, a1, false, b, (short)0, acc[1][j], false, false);
        }
    };

#if USE_ASYNC_LDS
    stage(0, 0);
    wait_async0();
    __syncthreads();
    int buf = 0;
#pragma unroll 1
    for (int k0 = BK; k0 < KD; k0 += BK) {
        stage(buf ^ 1, k0);   // async DMA next tile into other buffer...
        compute(buf);         // ...while WMMAs run on current buffer
        wait_async0();
        __syncthreads();
        buf ^= 1;
    }
    compute(buf);
#else
    g_load(0);
    l_store(0);
    __syncthreads();
    int buf = 0;
#pragma unroll 1
    for (int k0 = BK; k0 < KD; k0 += BK) {
        g_load(k0);
        compute(buf);
        l_store(buf ^ 1);
        __syncthreads();
        buf ^= 1;
    }
    compute(buf);
#endif
}

// ---------------- GEMM1: state = tanh(Xh @ Bh^T), bf16 out ---------------
__global__ __launch_bounds__(256)
void k_gemm1_tanh(const bf16_t* __restrict__ xh, const bf16_t* __restrict__ Bh,
                  bf16_t* __restrict__ state) {
    __shared__ __align__(16) bf16_t sA[2 * BM * LDSS];
    __shared__ __align__(16) bf16_t sB[2 * BN * LDSS];

    const int n0 = blockIdx.x * BN;       // over DSZ
    const int m0 = blockIdx.y * BM;

    v8f acc[2][4] = {};
    gemm_core<DM>(xh, Bh, m0, n0, sA, sB, acc);

    const int lane = threadIdx.x & 31;
    const int wave = threadIdx.x >> 5;
    const int wm2  = (wave & 3) * 2;
    const int wn4  = (wave >> 2) * 4;
    const int rr   = (lane >> 4) * 8;     // D layout: M = r + 8*(lane/16)
    const int cc   = lane & 15;           //           N = lane%16
#pragma unroll
    for (int i = 0; i < 2; ++i)
#pragma unroll
        for (int j = 0; j < 4; ++j)
#pragma unroll
            for (int r = 0; r < 8; ++r) {
                const int gm = m0 + (wm2 + i) * 16 + r + rr;
                const int gn = n0 + (wn4 + j) * 16 + cc;
                state[(size_t)gm * DSZ + gn] = f2bf(tanhf(acc[i][j][r]));
            }
}

// --------------- GEMM2: out = state @ Ch^T + D*x, fp32 out ---------------
__global__ __launch_bounds__(256)
void k_gemm2_out(const bf16_t* __restrict__ st, const bf16_t* __restrict__ Ch,
                 const float* __restrict__ Dv, const float* __restrict__ x,
                 float* __restrict__ out) {
    __shared__ __align__(16) bf16_t sA[2 * BM * LDSS];
    __shared__ __align__(16) bf16_t sB[2 * BN * LDSS];

    const int n0 = blockIdx.x * BN;       // over DM
    const int m0 = blockIdx.y * BM;

    v8f acc[2][4] = {};
    gemm_core<DSZ>(st, Ch, m0, n0, sA, sB, acc);

    const int lane = threadIdx.x & 31;
    const int wave = threadIdx.x >> 5;
    const int wm2  = (wave & 3) * 2;
    const int wn4  = (wave >> 2) * 4;
    const int rr   = (lane >> 4) * 8;
    const int cc   = lane & 15;
#pragma unroll
    for (int i = 0; i < 2; ++i)
#pragma unroll
        for (int j = 0; j < 4; ++j)
#pragma unroll
            for (int r = 0; r < 8; ++r) {
                const int gm = m0 + (wm2 + i) * 16 + r + rr;
                const int gn = n0 + (wn4 + j) * 16 + cc;
                const size_t o = (size_t)gm * DM + gn;
                out[o] = acc[i][j][r] + Dv[gn] * x[o];
            }
}

extern "C" void kernel_launch(void* const* d_in, const int* in_sizes, int n_in,
                              void* d_out, int out_size, void* d_ws, size_t ws_size,
                              hipStream_t stream) {
    const float* x  = (const float*)d_in[0];   // [16384, 2048]
    const float* Bm = (const float*)d_in[1];   // [1024, 2048]
    const float* Cm = (const float*)d_in[2];   // [2048, 1024]
    const float* Dv = (const float*)d_in[3];   // [2048]
    float* out = (float*)d_out;                // [16384, 2048]

    // ws: xh | Bh | Ch | state   (~108 MB total)
    bf16_t* xh = (bf16_t*)d_ws;
    bf16_t* Bh = xh + (size_t)MTOT * DM;
    bf16_t* Ch = Bh + (size_t)DSZ * DM;
    bf16_t* st = Ch + (size_t)DM * DSZ;

    const size_t nx = (size_t)MTOT * DM;       // 33.5M
    const size_t nb = (size_t)DSZ * DM;        // 2.1M
    const size_t nc = (size_t)DM * DSZ;        // 2.1M

    dim3 blk(256, 1, 1);
    k_pack_bf16<<<dim3((unsigned)(nx / 1024)), blk, 0, stream>>>(x,  xh, nx);
    k_pack_bf16<<<dim3((unsigned)(nb / 1024)), blk, 0, stream>>>(Bm, Bh, nb);
    k_pack_bf16<<<dim3((unsigned)(nc / 1024)), blk, 0, stream>>>(Cm, Ch, nc);

    k_gemm1_tanh<<<dim3(DSZ / BN, MTOT / BM), blk, 0, stream>>>(xh, Bh, st);
    k_gemm2_out <<<dim3(DM  / BN, MTOT / BM), blk, 0, stream>>>(st, Ch, Dv, x, out);
    (void)in_sizes; (void)n_in; (void)out_size; (void)ws_size;
}
